// DeepConvLSTM2_28312424415340
// MI455X (gfx1250) — matrix operand order repaired and checked by reading, compiled, not run on hardware
//
#include <hip/hip_runtime.h>
#include <hip/hip_bf16.h>

// ---------------------------------------------------------------------------
// DeepConvLSTM2 for gfx1250 (wave32, WMMA).  All matmuls via
// v_wmma_f32_16x16x32_bf16; B panels staged into LDS (async global->LDS when
// the toolchain exposes the gfx1250 builtin), transposed in LDS so fragments
// load as contiguous ds_load_b128 pairs; A fragments = two global_load_b128.
// ---------------------------------------------------------------------------

typedef __attribute__((ext_vector_type(16))) __bf16 v16bf;
typedef __attribute__((ext_vector_type(8)))  __bf16 v8bf;
typedef __attribute__((ext_vector_type(8)))  float  v8f;
typedef __attribute__((ext_vector_type(4)))  int    v4i;

#define SAMPLES 512
#define HLEN    128                // conv H dimension ("CHANNELS" in reference)
#define NH      (SAMPLES * HLEN)   // 65536 rows of the implicit GEMMs
#define HID     128
#define OUTN    6

#if defined(__HIP_DEVICE_COMPILE__) && __has_builtin(__builtin_amdgcn_global_load_async_to_lds_b128)
#define ASYNC_LDS 1
#endif

#ifdef ASYNC_LDS
typedef __attribute__((address_space(1))) v4i g_v4i;   // global 16B chunk
typedef __attribute__((address_space(3))) v4i l_v4i;   // LDS 16B chunk

__device__ __forceinline__ void async_copy16(const __bf16* g, __bf16* l) {
  __builtin_amdgcn_global_load_async_to_lds_b128(
      (g_v4i*)(void*)const_cast<__bf16*>(g),
      (l_v4i*)(void*)l, 0, 0);
}
__device__ __forceinline__ void wait_async() {
#if __has_builtin(__builtin_amdgcn_s_wait_asynccnt)
  __builtin_amdgcn_s_wait_asynccnt(0);
#else
  asm volatile("s_wait_asynccnt 0x0" ::: "memory");
#endif
}
#endif

// ---------------- WMMA fragment helpers (CDNA5 16x16x32 bf16) ---------------

__device__ __forceinline__ v16bf combine8(v8bf lo, v8bf hi) {
  v16bf r;
#pragma unroll
  for (int i = 0; i < 8; ++i) { r[i] = lo[i]; r[8 + i] = hi[i]; }
  return r;
}

// A fragment 16x32: lane L holds row M=L%16; element e -> K = e + (e>=8 ? 8:0)
// + 8*(L>=16): two contiguous 8-element chunks at +0 and +16 -> two 16B loads.
__device__ __forceinline__ v16bf load_a_frag(const __bf16* rowp, bool valid) {
  const int lane = threadIdx.x & 31;
  const __bf16* p = rowp + (((lane >> 4) & 1) << 3);
  const v8bf z = {};
  v8bf lo = valid ? *(const v8bf*)(p)      : z;
  v8bf hi = valid ? *(const v8bf*)(p + 16) : z;
  return combine8(lo, hi);
}

// B fragment 32x16 from the transposed LDS panel ldsT[n*Kpad + k]:
// lane L holds col N=L%16; K = kb..kb+7 and kb+16..kb+23 (kb=k0+8*(L>=16)) -> two 16B ds loads.
__device__ __forceinline__ v16bf load_b_frag_lds(const __bf16* ldsT, int Kpad,
                                                 int colbase, int k0) {
  const int lane = threadIdx.x & 31;
  const int n  = colbase + (lane & 15);
  const int kb = k0 + ((lane >> 4) << 3);
  const __bf16* p = ldsT + n * Kpad + kb;
  v8bf lo = *(const v8bf*)(p);
  v8bf hi = *(const v8bf*)(p + 16);
  return combine8(lo, hi);
}

__device__ __forceinline__ v8f wmma_bf16(v16bf a, v16bf b, v8f c) {
  return __builtin_amdgcn_wmma_f32_16x16x32_bf16(false, a, false, b, (short)0, c,
                                                 false, false);
}

// Stage a [KP rows x 16 cols] panel of B (row stride ldb, col offset c0) into
// LDS transposed as panel[n*Kpad + k].  raw = KP*16 scratch region (async path).
__device__ __forceinline__ void stage_panel(const __bf16* __restrict__ B, long ldb,
                                            int c0, int KP, int Kpad,
                                            __bf16* raw, __bf16* panel) {
#ifdef ASYNC_LDS
  for (int j = threadIdx.x; j < KP * 2; j += 256) {
    int kk = j >> 1, hh = (j & 1) << 3;
    async_copy16(B + (long)kk * ldb + c0 + hh, raw + kk * 16 + hh);
  }
  wait_async();
  __syncthreads();
  for (int i = threadIdx.x; i < KP * 16; i += 256) {
    int kk = i >> 4, n = i & 15;
    panel[n * Kpad + kk] = raw[kk * 16 + n];
  }
#else
  (void)raw;
  for (int i = threadIdx.x; i < KP * 16; i += 256) {
    int kk = i >> 4, n = i & 15;
    panel[n * Kpad + kk] = B[(long)kk * ldb + c0 + n];
  }
#endif
  __syncthreads();
}

// ---------------------------- utility kernels -------------------------------

__global__ void k_f32_to_bf16(const float* __restrict__ in, __bf16* __restrict__ out, int n) {
  int i = blockIdx.x * 256 + threadIdx.x;
  if (i < n) out[i] = (__bf16)in[i];
}

__global__ void k_zero_u32(unsigned int* __restrict__ p, int n) {
  int i = blockIdx.x * 256 + threadIdx.x;
  if (i < n) p[i] = 0u;
}

// Pack conv weight (5,5,Cin,Cout) f32 -> [5*Cin, Cout] bf16 keeping only kw=2
// (W=1 with SAME padding: only the center column of the 5x5 kernel is live).
__global__ void k_pack_convw(const float* __restrict__ w, __bf16* __restrict__ out,
                             int Cin, int Cout) {
  int i = blockIdx.x * 256 + threadIdx.x;
  int total = 5 * Cin * Cout;
  if (i >= total) return;
  int cout = i % Cout;
  int rest = i / Cout;
  int cin  = rest % Cin;
  int tap  = rest / Cin;
  long src = ((long)((tap * 5) + 2) * Cin + cin) * Cout + cout;
  out[i] = (__bf16)w[src];
}

// ------------------------- conv-as-GEMM (WMMA) -------------------------------
// out[NH,Cout] = relu(bias + sum_tap X[(n,h+tap-2), :Cin] x W[tap*Cin:, :Cout]).
// Block = 256 threads = 8 waves = one full sample (128 rows) x 16 channels.
// grid = (NH/128, Cout/16).  Dynamic LDS: raw[KP*16] + panel[16*Kpad].
__global__ void k_conv_wmma(const __bf16* __restrict__ A, const __bf16* __restrict__ W,
                            const float* __restrict__ bias, __bf16* __restrict__ out,
                            int Cin, int Cout, int Kpad) {
  extern __shared__ __bf16 smem[];
  const int KP = 5 * Cin;
  const int c0 = blockIdx.y << 4;
  __bf16* raw   = smem;
  __bf16* panel = smem + KP * 16;

  stage_panel(W, Cout, c0, KP, Kpad, raw, panel);

  const int w    = threadIdx.x >> 5;             // wave id = row tile in sample
  const int lane = threadIdx.x & 31;
  const int nbase = blockIdx.x << 7;             // n * 128
  const int h0    = w << 4;
  const int r0    = nbase + h0;
  const int m     = lane & 15;

  v8f acc = {};
  for (int tap = 0; tap < 5; ++tap) {
    const int hp = h0 + m + tap - 2;
    const bool valid = (hp >= 0) && (hp < HLEN);
    const __bf16* rowp = A + (long)(nbase + hp) * Cin;
    for (int k = 0; k < Cin; k += 32) {
      v16bf a = load_a_frag(rowp + k, valid);
      v16bf b = load_b_frag_lds(panel, Kpad, 0, tap * Cin + k);
      acc = wmma_bf16(a, b, acc);
    }
  }
  const int col = c0 + (lane & 15);
  const int mh  = (lane >> 4) << 3;
  const float bv = bias[col];
#pragma unroll
  for (int r = 0; r < 8; ++r) {
    float v = acc[r] + bv;
    out[(long)(r0 + mh + r) * Cout + col] = (__bf16)(v > 0.f ? v : 0.f);
  }
}

// --------------------- GEMM + bias (f32 out): xp = A@B + b -------------------
// Block = 8 waves = 128 rows x 16 cols; grid = (M/128, N/16).
__global__ void k_gemm_bias_wmma(const __bf16* __restrict__ A, const __bf16* __restrict__ B,
                                 const float* __restrict__ bias, float* __restrict__ out,
                                 int K, int N, int Kpad) {
  extern __shared__ __bf16 smem[];
  const int c0 = blockIdx.y << 4;
  __bf16* raw   = smem;
  __bf16* panel = smem + K * 16;

  stage_panel(B, N, c0, K, Kpad, raw, panel);

  const int w    = threadIdx.x >> 5;
  const int lane = threadIdx.x & 31;
  const int r0   = (blockIdx.x << 7) + (w << 4);
  const int m    = lane & 15;

  v8f acc = {};
  const __bf16* arow = A + (long)(r0 + m) * K;
  for (int k = 0; k < K; k += 32) {
    v16bf a = load_a_frag(arow + k, true);
    v16bf b = load_b_frag_lds(panel, Kpad, 0, k);
    acc = wmma_bf16(a, b, acc);
  }
  const int col = c0 + (lane & 15);
  const int mh  = (lane >> 4) << 3;
  const float bv = bias[col];
#pragma unroll
  for (int r = 0; r < 8; ++r)
    out[(long)(r0 + mh + r) * N + col] = acc[r] + bv;
}

// ------------------------------ LSTM step ------------------------------------
// z = xp_t + h_prev @ wh ; i,f,g,o gates ; c = f*c + i*g ; h = o*tanh(c).
// Block = 8 waves = 128 batch rows x 16 hidden cols; grid = (B/128, H/16) = (4,8).
// wh panel for all 4 gates (64 cols x K=128) staged transposed in static LDS.
#define LKPAD (HID + 8)   // 136; dword stride 68 % 64 == 4 -> conflict-friendly
__global__ void k_lstm_step_wmma(const float* __restrict__ xp_t, long xp_stride,
                                 const __bf16* __restrict__ wh,
                                 const __bf16* __restrict__ h_prev,
                                 float* __restrict__ c,
                                 __bf16* __restrict__ h_next,
                                 float* __restrict__ h_raw,
                                 __bf16* __restrict__ seq_out, long seq_stride) {
  __shared__ __bf16 raw[HID * 64];        // [k][64] gate-blocked raw panel
  __shared__ __bf16 pan[64 * LKPAD];      // [col][k] transposed panel
  const int H = HID;
  const int c0 = blockIdx.y << 4;

#ifdef ASYNC_LDS
  // chunk j: k = j>>3, group g = (j&7)>>1, half hh = j&1  (16B = 8 cols each)
  for (int j = threadIdx.x; j < H * 8; j += 256) {
    int k = j >> 3, g = (j & 7) >> 1, hh = (j & 1) << 3;
    async_copy16(wh + (long)k * (4 * H) + (g << 7) + c0 + hh,
                 raw + k * 64 + (g << 4) + hh);
  }
  wait_async();
  __syncthreads();
  for (int i = threadIdx.x; i < 64 * H; i += 256) {
    int k = i >> 6, col = i & 63;
    pan[col * LKPAD + k] = raw[k * 64 + col];
  }
#else
  for (int i = threadIdx.x; i < 64 * H; i += 256) {
    int k   = i >> 6;
    int col = i & 63;                                  // gate*16 + n
    int gcol = ((col >> 4) << 7) + c0 + (col & 15);    // gate*128 + c0 + n
    pan[col * LKPAD + k] = wh[(long)k * (4 * H) + gcol];
  }
#endif
  __syncthreads();

  const int w    = threadIdx.x >> 5;
  const int lane = threadIdx.x & 31;
  const int r0   = (blockIdx.x << 7) + (w << 4);
  const int m    = lane & 15;

  v8f a0 = {}, a1 = {}, a2 = {}, a3 = {};
  const __bf16* arow = h_prev + (long)(r0 + m) * H;
  for (int k = 0; k < H; k += 32) {
    v16bf a  = load_a_frag(arow + k, true);
    v16bf b0 = load_b_frag_lds(pan, LKPAD,  0, k);
    v16bf b1 = load_b_frag_lds(pan, LKPAD, 16, k);
    v16bf b2 = load_b_frag_lds(pan, LKPAD, 32, k);
    v16bf b3 = load_b_frag_lds(pan, LKPAD, 48, k);
    a0 = wmma_bf16(a, b0, a0);
    a1 = wmma_bf16(a, b1, a1);
    a2 = wmma_bf16(a, b2, a2);
    a3 = wmma_bf16(a, b3, a3);
  }
  const int col = c0 + (lane & 15);
  const int mh  = (lane >> 4) << 3;
#pragma unroll
  for (int r = 0; r < 8; ++r) {
    const int row = r0 + mh + r;
    const float* xr = xp_t + (long)row * xp_stride;
    float zi = a0[r] + xr[0 * H + col];
    float zf = a1[r] + xr[1 * H + col];
    float zg = a2[r] + xr[2 * H + col];
    float zo = a3[r] + xr[3 * H + col];
    float ig = 1.f / (1.f + __expf(-zi));
    float fg = 1.f / (1.f + __expf(-zf));
    float gg = tanhf(zg);
    float og = 1.f / (1.f + __expf(-zo));
    long idx = (long)row * H + col;
    float cn = fg * c[idx] + ig * gg;
    c[idx] = cn;
    float h = og * tanhf(cn);
    h_next[idx] = (__bf16)h;
    if (h_raw)  h_raw[idx] = h;
    if (seq_out) seq_out[(long)row * seq_stride + col] = (__bf16)(h > 0.f ? h : 0.f);
  }
}

// ------------------------------- dense head ----------------------------------
__global__ void k_dense_sigmoid(const float* __restrict__ h, const float* __restrict__ w,
                                const float* __restrict__ b, float* __restrict__ out) {
  int i = blockIdx.x * 256 + threadIdx.x;
  if (i >= SAMPLES * OUTN) return;
  int o = i % OUTN, row = i / OUTN;
  float s = b[o];
  for (int k = 0; k < HID; ++k) {
    float hv = h[(long)row * HID + k];
    hv = hv > 0.f ? hv : 0.f;
    s += hv * w[k * OUTN + o];
  }
  out[i] = 1.f / (1.f + __expf(-s));
}

// ------------------------------- launcher ------------------------------------

static inline size_t ws_alloc(size_t* off, size_t bytes) {
  size_t o = *off;
  *off = (*off + bytes + 255) & ~(size_t)255;
  return o;
}

// LDS bytes for a staged panel kernel: raw[KP*16] + panel[16*(KP+8)], bf16.
static inline size_t panel_lds(int KP) {
  return ((size_t)KP * 16 + (size_t)(KP + 8) * 16) * 2;
}

extern "C" void kernel_launch(void* const* d_in, const int* in_sizes, int n_in,
                              void* d_out, int out_size, void* d_ws, size_t ws_size,
                              hipStream_t stream) {
  (void)in_sizes; (void)n_in; (void)out_size; (void)ws_size;
  const float* x    = (const float*)d_in[0];
  const float* c1w  = (const float*)d_in[1];
  const float* c1b  = (const float*)d_in[2];
  const float* c2w  = (const float*)d_in[3];
  const float* c2b  = (const float*)d_in[4];
  const float* c3w  = (const float*)d_in[5];
  const float* c3b  = (const float*)d_in[6];
  const float* c4w  = (const float*)d_in[7];
  const float* c4b  = (const float*)d_in[8];
  const float* l1wx = (const float*)d_in[9];
  const float* l1wh = (const float*)d_in[10];
  const float* l1b  = (const float*)d_in[11];
  const float* l2wx = (const float*)d_in[12];
  const float* l2wh = (const float*)d_in[13];
  const float* l2b  = (const float*)d_in[14];
  const float* dw   = (const float*)d_in[15];
  const float* db   = (const float*)d_in[16];
  float* out        = (float*)d_out;
  char* ws          = (char*)d_ws;

  size_t off = 0;
  __bf16* xbf   = (__bf16*)(ws + ws_alloc(&off, (size_t)NH * 64 * 2));
  __bf16* a1    = (__bf16*)(ws + ws_alloc(&off, (size_t)NH * 64 * 2));
  __bf16* a2    = (__bf16*)(ws + ws_alloc(&off, (size_t)NH * 128 * 2));
  __bf16* a3    = (__bf16*)(ws + ws_alloc(&off, (size_t)NH * 256 * 2));
  __bf16* a4    = (__bf16*)(ws + ws_alloc(&off, (size_t)NH * 512 * 2));
  __bf16* w1p   = (__bf16*)(ws + ws_alloc(&off, (size_t)5 * 64 * 64 * 2));
  __bf16* w2p   = (__bf16*)(ws + ws_alloc(&off, (size_t)5 * 64 * 128 * 2));
  __bf16* w3p   = (__bf16*)(ws + ws_alloc(&off, (size_t)5 * 128 * 256 * 2));
  __bf16* w4p   = (__bf16*)(ws + ws_alloc(&off, (size_t)5 * 256 * 512 * 2));
  __bf16* wx1   = (__bf16*)(ws + ws_alloc(&off, (size_t)512 * 512 * 2));
  __bf16* wh1   = (__bf16*)(ws + ws_alloc(&off, (size_t)128 * 512 * 2));
  __bf16* wx2   = (__bf16*)(ws + ws_alloc(&off, (size_t)128 * 512 * 2));
  __bf16* wh2   = (__bf16*)(ws + ws_alloc(&off, (size_t)128 * 512 * 2));
  float*  xp    = (float*) (ws + ws_alloc(&off, (size_t)NH * 512 * 4)); // lstm1+lstm2
  __bf16* h1seq = (__bf16*)(ws + ws_alloc(&off, (size_t)NH * 128 * 2));
  __bf16* hbufA = (__bf16*)(ws + ws_alloc(&off, (size_t)SAMPLES * HID * 2));
  __bf16* hbufB = (__bf16*)(ws + ws_alloc(&off, (size_t)SAMPLES * HID * 2));
  float*  cbuf  = (float*) (ws + ws_alloc(&off, (size_t)SAMPLES * HID * 4));
  float*  hraw  = (float*) (ws + ws_alloc(&off, (size_t)SAMPLES * HID * 4));

  const dim3 B256(256);
  auto grid1d = [](int n) { return dim3((unsigned)((n + 255) / 256)); };

  // --- precision conversion / weight packing ---
  k_f32_to_bf16<<<grid1d(NH * 64), B256, 0, stream>>>(x, xbf, NH * 64);
  k_pack_convw<<<grid1d(5 * 64 * 64),   B256, 0, stream>>>(c1w, w1p, 64, 64);
  k_pack_convw<<<grid1d(5 * 64 * 128),  B256, 0, stream>>>(c2w, w2p, 64, 128);
  k_pack_convw<<<grid1d(5 * 128 * 256), B256, 0, stream>>>(c3w, w3p, 128, 256);
  k_pack_convw<<<grid1d(5 * 256 * 512), B256, 0, stream>>>(c4w, w4p, 256, 512);
  k_f32_to_bf16<<<grid1d(512 * 512), B256, 0, stream>>>(l1wx, wx1, 512 * 512);
  k_f32_to_bf16<<<grid1d(128 * 512), B256, 0, stream>>>(l1wh, wh1, 128 * 512);
  k_f32_to_bf16<<<grid1d(128 * 512), B256, 0, stream>>>(l2wx, wx2, 128 * 512);
  k_f32_to_bf16<<<grid1d(128 * 512), B256, 0, stream>>>(l2wh, wh2, 128 * 512);

  // --- conv stack: grid (NH/128, Cout/16) ---
  k_conv_wmma<<<dim3(NH / 128, 64 / 16),  B256, panel_lds(5 * 64),  stream>>>(xbf, w1p, c1b, a1, 64, 64,  5 * 64  + 8);
  k_conv_wmma<<<dim3(NH / 128, 128 / 16), B256, panel_lds(5 * 64),  stream>>>(a1,  w2p, c2b, a2, 64, 128, 5 * 64  + 8);
  k_conv_wmma<<<dim3(NH / 128, 256 / 16), B256, panel_lds(5 * 128), stream>>>(a2,  w3p, c3b, a3, 128, 256, 5 * 128 + 8);
  k_conv_wmma<<<dim3(NH / 128, 512 / 16), B256, panel_lds(5 * 256), stream>>>(a3,  w4p, c4b, a4, 256, 512, 5 * 256 + 8);

  // --- LSTM1: xp = a4 @ wx1 + b1, then 128 recurrent steps ---
  k_gemm_bias_wmma<<<dim3(NH / 128, 512 / 16), B256, panel_lds(512), stream>>>(
      a4, wx1, l1b, xp, 512, 512, 512 + 8);

  k_zero_u32<<<grid1d(SAMPLES * HID), B256, 0, stream>>>((unsigned int*)cbuf, SAMPLES * HID);
  k_zero_u32<<<grid1d(SAMPLES * HID / 2), B256, 0, stream>>>((unsigned int*)hbufA, SAMPLES * HID / 2);

  const dim3 step_grid(SAMPLES / 128, HID / 16);   // (4, 8)
  __bf16* hping[2] = {hbufA, hbufB};
  for (int t = 0; t < HLEN; ++t) {
    k_lstm_step_wmma<<<step_grid, B256, 0, stream>>>(
        xp + (long)t * 512, (long)HLEN * 512, wh1,
        hping[t & 1], cbuf, hping[(t + 1) & 1],
        (float*)nullptr, h1seq + (long)t * 128, (long)HLEN * 128);
  }

  // --- LSTM2: xp = relu(h1seq) @ wx2 + b2 (relu already applied), 128 steps ---
  k_gemm_bias_wmma<<<dim3(NH / 128, 512 / 16), B256, panel_lds(128), stream>>>(
      h1seq, wx2, l2b, xp, 128, 512, 128 + 8);

  k_zero_u32<<<grid1d(SAMPLES * HID), B256, 0, stream>>>((unsigned int*)cbuf, SAMPLES * HID);
  k_zero_u32<<<grid1d(SAMPLES * HID / 2), B256, 0, stream>>>((unsigned int*)hbufA, SAMPLES * HID / 2);

  for (int t = 0; t < HLEN; ++t) {
    k_lstm_step_wmma<<<step_grid, B256, 0, stream>>>(
        xp + (long)t * 512, (long)HLEN * 512, wh2,
        hping[t & 1], cbuf, hping[(t + 1) & 1],
        hraw, (__bf16*)nullptr, 0);
  }

  // --- dense + sigmoid ---
  k_dense_sigmoid<<<grid1d(SAMPLES * OUTN), B256, 0, stream>>>(hraw, dw, db, out);
}
